// LinearAttention_80985903333478
// MI455X (gfx1250) — matrix operand order repaired and checked by reading, not compiled, not run on hardware
//
#include <hip/hip_runtime.h>
#include <math.h>

typedef _Float16 f16;
typedef _Float16 v16h __attribute__((ext_vector_type(16)));
typedef float    v8f  __attribute__((ext_vector_type(8)));

#define B_ 4
#define T_ 8192
#define C_ 1024
#define H_ 16
#define MTOT (B_*T_)
#define SPLIT 8

// LDS byte offset of a __shared__ element (ptrtoint of an AS(3) pointer).
#define LDS_OFF(p) ((unsigned)(uintptr_t)((__attribute__((address_space(3))) const void*)(p)))

static __device__ __forceinline__ v8f wmma_f16(v16h a, v16h b, v8f c) {
  return __builtin_amdgcn_wmma_f32_16x16x32_f16(false, a, false, b, (short)0, c, false, false);
}

static __device__ __forceinline__ v8f vzero8() {
  v8f z = {0.f,0.f,0.f,0.f,0.f,0.f,0.f,0.f};
  return z;
}

// Async copy of 16 bytes global -> LDS (per-lane addresses), ASYNCcnt-tracked.
static __device__ __forceinline__ void async_b128(unsigned lds_byte_off, const void* gptr) {
  asm volatile("global_load_async_to_lds_b128 %0, %1, off"
               :: "v"(lds_byte_off), "v"(gptr) : "memory");
}
static __device__ __forceinline__ void wait_async0() {
  asm volatile("s_wait_asynccnt 0x0" ::: "memory");
}

// ---------------------------------------------------------------------------
// K1: fused QKV projection.  y = x @ W^T + b  (W is [N=1024, K=1024] row-major,
// so both x rows and W rows are contiguous over K).  blockIdx.z picks q/k/v.
// Tile: BM=128 (8 waves x 16 rows), BN=64, BK=32.  Output f16 (B,T,C).
// (fp32 -> f16 conversion happens in the VGPR path, so no async copy here.)
// ---------------------------------------------------------------------------
__global__ __launch_bounds__(256) void k_qkv_gemm(
    const float* __restrict__ x,
    const float* __restrict__ Wq, const float* __restrict__ bq, f16* __restrict__ qo,
    const float* __restrict__ Wk, const float* __restrict__ bk, f16* __restrict__ ko,
    const float* __restrict__ Wv, const float* __restrict__ bv, f16* __restrict__ vo)
{
  const float* W; const float* bias; f16* out;
  if      (blockIdx.z == 0) { W = Wq; bias = bq; out = qo; }
  else if (blockIdx.z == 1) { W = Wk; bias = bk; out = ko; }
  else                      { W = Wv; bias = bv; out = vo; }

  __shared__ __align__(16) f16 Asm[128][32];
  __shared__ __align__(16) f16 Bsm[64][32];

  const int tid = threadIdx.x;
  const int w = tid >> 5, lane = tid & 31;
  const int lm = lane & 15, hi = lane >> 4;
  const int m0 = blockIdx.x * 128, n0 = blockIdx.y * 64;

  v8f acc[4];
  #pragma unroll
  for (int i = 0; i < 4; ++i) acc[i] = vzero8();

  const int arow = tid >> 1, acol = (tid & 1) * 16;
  const int brow = tid >> 2, bcol = (tid & 3) * 8;

  for (int k0 = 0; k0 < C_; k0 += 32) {
    {
      const float4* s = reinterpret_cast<const float4*>(x + (size_t)(m0 + arow) * C_ + k0 + acol);
      if (k0 + 64 < C_)
        __builtin_prefetch(x + (size_t)(m0 + arow) * C_ + k0 + 64 + acol, 0, 1);
      #pragma unroll
      for (int i = 0; i < 4; ++i) {
        float4 f = s[i];
        Asm[arow][acol + 4*i + 0] = (f16)f.x; Asm[arow][acol + 4*i + 1] = (f16)f.y;
        Asm[arow][acol + 4*i + 2] = (f16)f.z; Asm[arow][acol + 4*i + 3] = (f16)f.w;
      }
    }
    {
      const float4* s = reinterpret_cast<const float4*>(W + (size_t)(n0 + brow) * C_ + k0 + bcol);
      #pragma unroll
      for (int i = 0; i < 2; ++i) {
        float4 f = s[i];
        Bsm[brow][bcol + 4*i + 0] = (f16)f.x; Bsm[brow][bcol + 4*i + 1] = (f16)f.y;
        Bsm[brow][bcol + 4*i + 2] = (f16)f.z; Bsm[brow][bcol + 4*i + 3] = (f16)f.w;
      }
    }
    __syncthreads();

    v16h a;
    {
      const int am = 16 * w + lm;
      const int hb = hi * 16;
      #pragma unroll
      for (int j = 0; j < 8; ++j) a[j]     = Asm[am][hb + j];
      #pragma unroll
      for (int j = 0; j < 8; ++j) a[8 + j] = Asm[am][hb + 8 + j];
    }
    #pragma unroll
    for (int nt = 0; nt < 4; ++nt) {
      v16h b;
      const int bn = 16 * nt + lm;
      const int bk2 = hi * 16;
      #pragma unroll
      for (int j = 0; j < 16; ++j) b[j] = Bsm[bn][bk2 + j];
      acc[nt] = wmma_f16(a, b, acc[nt]);
    }
    __syncthreads();
  }

  #pragma unroll
  for (int nt = 0; nt < 4; ++nt) {
    #pragma unroll
    for (int r = 0; r < 8; ++r) {
      const int m = m0 + 16 * w + r + 8 * hi;
      const int n = n0 + 16 * nt + lm;
      out[(size_t)m * C_ + n] = (f16)(acc[nt][r] + bias[n]);
    }
  }
}

// ---------------------------------------------------------------------------
// K2: softmax over the 64-wide head dim, in place, one row per thread.
// Rows (b,t,h) are contiguous 64-element slices.  grid.y: 0 -> q, 1 -> k.
// ---------------------------------------------------------------------------
__global__ __launch_bounds__(256) void k_softmax64(f16* __restrict__ q, f16* __restrict__ k)
{
  f16* buf = (blockIdx.y == 0) ? q : k;
  const size_t row = (size_t)blockIdx.x * blockDim.x + threadIdx.x;
  f16* p = buf + row * 64;
  float v[64];
  float mx = -1e30f;
  #pragma unroll
  for (int i = 0; i < 64; ++i) { v[i] = (float)p[i]; mx = fmaxf(mx, v[i]); }
  float s = 0.f;
  #pragma unroll
  for (int i = 0; i < 64; ++i) { v[i] = __expf(v[i] - mx); s += v[i]; }
  const float inv = 1.f / s;
  #pragma unroll
  for (int i = 0; i < 64; ++i) p[i] = (f16)(v[i] * inv);
}

// ---------------------------------------------------------------------------
// K3: k_sum[b,h,d] = sum_t k[b,t,h,d]; also zero-init the context accumulator.
// One block per (b,h).
// ---------------------------------------------------------------------------
__global__ __launch_bounds__(256) void k_ksum_zeroctx(const f16* __restrict__ kbuf,
                                                      float* __restrict__ ksum,
                                                      float* __restrict__ ctx)
{
  const int bh = blockIdx.x, tid = threadIdx.x;
  for (int i = tid; i < 4096; i += 256) ctx[(size_t)bh * 4096 + i] = 0.f;
  const int d = tid & 63, part = tid >> 6;
  const int b = bh / H_, h = bh % H_;
  const f16* base = kbuf + ((size_t)b * T_) * C_ + h * 64 + d;
  float s = 0.f;
  for (int t = part; t < T_; t += 4) s += (float)base[(size_t)t * C_];
  __shared__ float red[256];
  red[tid] = s;
  __syncthreads();
  if (tid < 64)
    ksum[bh * 64 + tid] = red[tid] + red[tid + 64] + red[tid + 128] + red[tid + 192];
}

// ---------------------------------------------------------------------------
// K4: context[b,h,d,e] = sum_t k[b,t,h,d] * v[b,t,h,e]   (64x64, K = T).
// T split SPLIT-ways across blocks; partials reduced with f32 atomics.
// Tile fills are pure f16 copies -> use GLOBAL_LOAD_ASYNC_TO_LDS_B128.
// ---------------------------------------------------------------------------
__global__ __launch_bounds__(256) void k_context(const f16* __restrict__ kbuf,
                                                 const f16* __restrict__ vbuf,
                                                 float* __restrict__ ctx)
{
  const int bh = blockIdx.x / SPLIT, seg = blockIdx.x % SPLIT;
  const int b = bh / H_, h = bh % H_;
  const size_t base = ((size_t)b * T_) * C_ + h * 64;

  __shared__ __align__(16) f16 Ksm[32][64];
  __shared__ __align__(16) f16 Vsm[32][64];

  const int tid = threadIdx.x, w = tid >> 5, lane = tid & 31;
  const int lm = lane & 15, hi = lane >> 4;
  const int mi  = (2 * w) >> 2;          // tiles 2w and 2w+1 share mi
  const int ni0 = (2 * w) & 3, ni1 = ni0 + 1;

  v8f acc0 = vzero8(), acc1 = vzero8();

  const int frow = tid >> 3, fcol = (tid & 7) * 8;
  const unsigned ldsK = LDS_OFF(&Ksm[frow][fcol]);
  const unsigned ldsV = LDS_OFF(&Vsm[frow][fcol]);
  const int tBeg = seg * (T_ / SPLIT), tEnd = tBeg + (T_ / SPLIT);

  for (int t0 = tBeg; t0 < tEnd; t0 += 32) {
    const size_t goff = base + (size_t)(t0 + frow) * C_ + fcol;
    async_b128(ldsK, kbuf + goff);
    async_b128(ldsV, vbuf + goff);
    wait_async0();
    __syncthreads();

    v16h a;                                   // A = k^T : A[d][t]
    {
      const int m = mi * 16 + lm, hb = hi * 16;
      #pragma unroll
      for (int j = 0; j < 8; ++j) a[j]     = Ksm[hb + j][m];
      #pragma unroll
      for (int j = 0; j < 8; ++j) a[8 + j] = Ksm[hb + 8 + j][m];
    }
    {
      v16h bb;
      const int n = ni0 * 16 + lm;
      #pragma unroll
      for (int j = 0; j < 16; ++j) bb[j] = Vsm[hi * 16 + j][n];
      acc0 = wmma_f16(a, bb, acc0);
    }
    {
      v16h bb;
      const int n = ni1 * 16 + lm;
      #pragma unroll
      for (int j = 0; j < 16; ++j) bb[j] = Vsm[hi * 16 + j][n];
      acc1 = wmma_f16(a, bb, acc1);
    }
    __syncthreads();
  }

  #pragma unroll
  for (int r = 0; r < 8; ++r) {
    const int dd = mi * 16 + r + 8 * hi;
    atomicAdd(&ctx[(size_t)bh * 4096 + dd * 64 + ni0 * 16 + lm], acc0[r]);
    atomicAdd(&ctx[(size_t)bh * 4096 + dd * 64 + ni1 * 16 + lm], acc1[r]);
  }
}

// ---------------------------------------------------------------------------
// K5: out = (q @ context) * D_inv + q   per head, where
//     D_inv[t] = 1 / sum_d q[t,d] * k_sum[d].
// 128 rows per block (8 waves x 16), K = 64 (two WMMAs).  Output may alias the
// q buffer: all q data is staged through LDS before any global store.
// Q tile fill is a pure f16 copy -> async load to LDS.
// ---------------------------------------------------------------------------
__global__ __launch_bounds__(256) void k_attn_out(const f16* __restrict__ qbuf,
                                                  const float* __restrict__ ctx,
                                                  const float* __restrict__ ksum,
                                                  f16* __restrict__ yo)
{
  const int bh = blockIdx.y, b = bh / H_, h = bh % H_;
  const int t0 = blockIdx.x * 128;
  const size_t base = ((size_t)b * T_) * C_ + h * 64;

  __shared__ __align__(16) f16 Qsm[128][64];
  __shared__ __align__(16) f16 Bsm[64][64];   // context transposed: Bsm[e][d]
  __shared__ float dinv[128];
  __shared__ float ks[64];

  const int tid = threadIdx.x, w = tid >> 5, lane = tid & 31;
  const int lm = lane & 15, hi = lane >> 4;

  {
    const int row = tid >> 1, col = (tid & 1) * 32;
    const f16* gq = qbuf + base + (size_t)(t0 + row) * C_ + col;
    const unsigned ldsQ = LDS_OFF(&Qsm[row][col]);
    #pragma unroll
    for (int i = 0; i < 4; ++i)
      async_b128(ldsQ + 16u * i, gq + 8 * i);
  }
  {
    const float* cbase = ctx + (size_t)bh * 4096;
    #pragma unroll
    for (int i = 0; i < 16; ++i) {
      const int idx = tid * 16 + i;
      Bsm[idx & 63][idx >> 6] = (f16)cbase[idx];
    }
  }
  if (tid < 64) ks[tid] = ksum[bh * 64 + tid];
  wait_async0();
  __syncthreads();

  if (tid < 128) {
    float s = 0.f;
    #pragma unroll
    for (int d = 0; d < 64; ++d) s += (float)Qsm[tid][d] * ks[d];
    dinv[tid] = 1.f / s;
  }
  __syncthreads();

  v8f acc[4];
  #pragma unroll
  for (int i = 0; i < 4; ++i) acc[i] = vzero8();

  #pragma unroll
  for (int s = 0; s < 2; ++s) {
    v16h a;
    {
      const int am = 16 * w + lm, hb = s * 32 + hi * 16;
      #pragma unroll
      for (int j = 0; j < 8; ++j) a[j]     = Qsm[am][hb + j];
      #pragma unroll
      for (int j = 0; j < 8; ++j) a[8 + j] = Qsm[am][hb + 8 + j];
    }
    #pragma unroll
    for (int nt = 0; nt < 4; ++nt) {
      v16h bb;
      const int bn = nt * 16 + lm, bk = s * 32 + hi * 16;
      #pragma unroll
      for (int j = 0; j < 16; ++j) bb[j] = Bsm[bn][bk + j];
      acc[nt] = wmma_f16(a, bb, acc[nt]);
    }
  }

  #pragma unroll
  for (int nt = 0; nt < 4; ++nt) {
    #pragma unroll
    for (int r = 0; r < 8; ++r) {
      const int rowl = 16 * w + r + 8 * hi;
      const int e = nt * 16 + lm;
      const float vv = acc[nt][r] * dinv[rowl] + (float)Qsm[rowl][e];
      yo[base + (size_t)(t0 + rowl) * C_ + e] = (f16)vv;
    }
  }
}

// ---------------------------------------------------------------------------
// K6: final projection: out = y @ Wp^T + bp, f16 A, fp32 output.
// ---------------------------------------------------------------------------
__global__ __launch_bounds__(256) void k_proj_gemm(const f16* __restrict__ A,
                                                   const float* __restrict__ W,
                                                   const float* __restrict__ bias,
                                                   float* __restrict__ out)
{
  __shared__ __align__(16) f16 Asm[128][32];
  __shared__ __align__(16) f16 Bsm[64][32];

  const int tid = threadIdx.x;
  const int w = tid >> 5, lane = tid & 31;
  const int lm = lane & 15, hi = lane >> 4;
  const int m0 = blockIdx.x * 128, n0 = blockIdx.y * 64;

  v8f acc[4];
  #pragma unroll
  for (int i = 0; i < 4; ++i) acc[i] = vzero8();

  const int arow = tid >> 1, acol = (tid & 1) * 16;
  const int brow = tid >> 2, bcol = (tid & 3) * 8;
  const unsigned ldsA = LDS_OFF(&Asm[arow][acol]);

  for (int k0 = 0; k0 < C_; k0 += 32) {
    {
      // A tile is already f16: async copy (2 x 16B per thread).
      const f16* ga = A + (size_t)(m0 + arow) * C_ + k0 + acol;
      async_b128(ldsA,       ga);
      async_b128(ldsA + 16u, ga + 8);
    }
    {
      const float4* s = reinterpret_cast<const float4*>(W + (size_t)(n0 + brow) * C_ + k0 + bcol);
      #pragma unroll
      for (int i = 0; i < 2; ++i) {
        float4 f = s[i];
        Bsm[brow][bcol + 4*i + 0] = (f16)f.x; Bsm[brow][bcol + 4*i + 1] = (f16)f.y;
        Bsm[brow][bcol + 4*i + 2] = (f16)f.z; Bsm[brow][bcol + 4*i + 3] = (f16)f.w;
      }
    }
    wait_async0();
    __syncthreads();

    v16h a;
    {
      const int am = 16 * w + lm, hb = hi * 16;
      #pragma unroll
      for (int j = 0; j < 8; ++j) a[j]     = Asm[am][hb + j];
      #pragma unroll
      for (int j = 0; j < 8; ++j) a[8 + j] = Asm[am][hb + 8 + j];
    }
    #pragma unroll
    for (int nt = 0; nt < 4; ++nt) {
      v16h b;
      const int bn = 16 * nt + lm, bk2 = hi * 16;
      #pragma unroll
      for (int j = 0; j < 16; ++j) b[j] = Bsm[bn][bk2 + j];
      acc[nt] = wmma_f16(a, b, acc[nt]);
    }
    __syncthreads();
  }

  #pragma unroll
  for (int nt = 0; nt < 4; ++nt) {
    #pragma unroll
    for (int r = 0; r < 8; ++r) {
      const int m = m0 + 16 * w + r + 8 * hi;
      const int n = n0 + 16 * nt + lm;
      out[(size_t)m * C_ + n] = acc[nt][r] + bias[n];
    }
  }
}

// ---------------------------------------------------------------------------
extern "C" void kernel_launch(void* const* d_in, const int* in_sizes, int n_in,
                              void* d_out, int out_size, void* d_ws, size_t ws_size,
                              hipStream_t stream)
{
  const float* x  = (const float*)d_in[0];
  const float* Wq = (const float*)d_in[1];
  const float* bq = (const float*)d_in[2];
  const float* Wk = (const float*)d_in[3];
  const float* bk = (const float*)d_in[4];
  const float* Wv = (const float*)d_in[5];
  const float* bv = (const float*)d_in[6];
  const float* Wp = (const float*)d_in[7];
  const float* bp = (const float*)d_in[8];

  const size_t nE = (size_t)B_ * T_ * C_;
  f16*   qh   = (f16*)d_ws;        // 64 MB (post-K5: aliased as attention output y)
  f16*   kh   = qh + nE;           // 64 MB
  f16*   vh   = kh + nE;           // 64 MB
  float* ksum = (float*)(vh + nE); // 16 KB
  float* ctx  = ksum + (size_t)B_ * H_ * 64;  // 1 MB

  const dim3 blk(256);

  k_qkv_gemm<<<dim3(MTOT / 128, C_ / 64, 3), blk, 0, stream>>>(
      x, Wq, bq, qh, Wk, bk, kh, Wv, bv, vh);

  k_softmax64<<<dim3((unsigned)((size_t)B_ * T_ * H_ / 256), 2), blk, 0, stream>>>(qh, kh);

  k_ksum_zeroctx<<<dim3(B_ * H_), blk, 0, stream>>>(kh, ksum, ctx);

  k_context<<<dim3(B_ * H_ * SPLIT), blk, 0, stream>>>(kh, vh, ctx);

  k_attn_out<<<dim3(T_ / 128, B_ * H_), blk, 0, stream>>>(qh, ctx, ksum, qh /* alias */);

  k_proj_gemm<<<dim3(MTOT / 128, C_ / 64), blk, 0, stream>>>(qh, Wp, bp, (float*)d_out);
}
